// ModifiedCTMForIntervention_65352222376103
// MI455X (gfx1250) — matrix operand order repaired and checked by reading, not compiled, not run on hardware
//
#include <hip/hip_runtime.h>

// ---------------------------------------------------------------------------
// Model dimensions (from the reference)
// ---------------------------------------------------------------------------
namespace {
constexpr int B = 64, S = 64, DIN = 512, DATTN = 512, DMODEL = 2048;
constexpr int TTR = 25, HEADS = 8, HD = 64, NSYNC = 512, OUT = 1000, T = 50, HNLM = 32;
constexpr int PREMW = DATTN + DMODEL;   // 2560 : concat(ao, act)
constexpr int H2 = 2 * DMODEL;          // 4096 : Ws1 output
constexpr int OUTP = 1024;              // OUT padded to a multiple of 64
constexpr long PRED_ELEMS = (long)B * OUT * T;
constexpr long CERT_ELEMS = (long)B * 2 * T;
}

typedef __attribute__((ext_vector_type(4)))  float  f32x4;
typedef __attribute__((ext_vector_type(8)))  float  v8f;
typedef __attribute__((ext_vector_type(16))) __bf16 v16bf;
typedef unsigned short ushort_t;

static __device__ __forceinline__ ushort_t f32_to_bf16(float f) {
  union { float f; unsigned int u; } x; x.f = f;
  const unsigned int u = x.u;
  return (ushort_t)((u + 0x7FFFu + ((u >> 16) & 1u)) >> 16);  // RNE
}

// ---------------------------------------------------------------------------
// Weight conversion: f32 (K x N) row-major -> bf16 transposed (Npad x K).
// With this layout each WMMA B-fragment is one contiguous 32B run per lane.
// ---------------------------------------------------------------------------
__global__ void __launch_bounds__(256) k_convT(const float* __restrict__ W,
                                               ushort_t* __restrict__ WT,
                                               int K, int N, long total) {
  const long idx = (long)blockIdx.x * 256 + threadIdx.x;
  if (idx >= total) return;
  const int n = (int)(idx / K);
  const int k = (int)(idx - (long)n * K);
  const float v = (n < N) ? W[(size_t)k * N + n] : 0.0f;
  WT[idx] = f32_to_bf16(v);
}

__global__ void __launch_bounds__(256) k_conv(const float* __restrict__ in,
                                              ushort_t* __restrict__ outp, long n) {
  const long i = (long)blockIdx.x * 256 + threadIdx.x;
  if (i < n) outp[i] = f32_to_bf16(in[i]);
}

// ---------------------------------------------------------------------------
// bf16 WMMA GEMM:  C(M x N) = A(M x K, row-major bf16) * BT(N x K, bf16) + bias
//   - block = 128 threads (4 waves); block tile = 64(M) x 32(N)
//   - wave w owns rows [w*16, w*16+16) and 2 N-tiles (2 accumulators)
//   - ping-pong double buffering WITHOUT register rotation: the K loop is
//     unrolled 2x with two named buffers, so the WMMAs on one buffer overlap
//     the in-flight b128 loads of the other (partial s_wait_loadcnt), and no
//     v_mov copies / WMMA->VALU hazard NOPs are needed.
//   - requires K % 64 == 0 (all K in this model: 512 / 512 / 512 / 2560)
//   - f32 epilogue (optionally strided into d_out) and/or bf16 row-major out
// ---------------------------------------------------------------------------
__global__ void __launch_bounds__(128) k_gemm_bf16(
    const ushort_t* __restrict__ A, const ushort_t* __restrict__ BT,
    const float* __restrict__ bias,
    float* __restrict__ Cf, long crs, long ccs, long coff,
    ushort_t* __restrict__ Cb, int cb_ld,
    int K, int nmax) {
  const int lane = threadIdx.x & 31;
  const int wave = threadIdx.x >> 5;
  const int row0 = blockIdx.y * 64 + wave * 16;
  const int col0 = blockIdx.x * 32;

  const int m    = row0 + (lane & 15);
  const int kh8  = (lane >> 4) * 8;    // unused (A also uses kh16)
  const int kh16 = (lane >> 4) * 16;   // B K-half select
  const size_t arow  = (size_t)m * K;
  const size_t brow0 = (size_t)(col0 + (lane & 15)) * K;
  const size_t brow1 = brow0 + (size_t)16 * K;

  union U { f32x4 q[2]; v16bf v; };
  v8f acc0 = {}, acc1 = {};

#define LOAD_CHUNK(dst_a, dst_b0, dst_b1, kk)                         \
  do {                                                                \
    (dst_a).q[0]  = *(const f32x4*)(A + arow + (kk) + kh16);          \
    (dst_a).q[1]  = *(const f32x4*)(A + arow + (kk) + kh16 + 8);      \
    (dst_b0).q[0] = *(const f32x4*)(BT + brow0 + (kk) + kh16);        \
    (dst_b0).q[1] = *(const f32x4*)(BT + brow0 + (kk) + kh16 + 8);    \
    (dst_b1).q[0] = *(const f32x4*)(BT + brow1 + (kk) + kh16);        \
    (dst_b1).q[1] = *(const f32x4*)(BT + brow1 + (kk) + kh16 + 8);    \
  } while (0)

  U aP, b0P, b1P;   // ping
  U aQ, b0Q, b1Q;   // pong
  LOAD_CHUNK(aP, b0P, b1P, 0);

  for (int kc = 0; kc < K; kc += 64) {
    // issue pong loads, then consume ping (its loads are one phase older)
    LOAD_CHUNK(aQ, b0Q, b1Q, kc + 32);
    if (kc + 64 < K)  // keep a future chunk warm in L2 (global_prefetch_b8)
      __builtin_prefetch(BT + brow0 + kc + 64, 0, 3);
    acc0 = __builtin_amdgcn_wmma_f32_16x16x32_bf16(
        false, aP.v, false, b0P.v, (short)0, acc0, false, false);
    acc1 = __builtin_amdgcn_wmma_f32_16x16x32_bf16(
        false, aP.v, false, b1P.v, (short)0, acc1, false, false);
    // issue next ping loads, then consume pong
    if (kc + 64 < K) LOAD_CHUNK(aP, b0P, b1P, kc + 64);
    acc0 = __builtin_amdgcn_wmma_f32_16x16x32_bf16(
        false, aQ.v, false, b0Q.v, (short)0, acc0, false, false);
    acc1 = __builtin_amdgcn_wmma_f32_16x16x32_bf16(
        false, aQ.v, false, b1Q.v, (short)0, acc1, false, false);
  }
#undef LOAD_CHUNK

  // C/D layout: VGPR r -> row = r + 8*(lane>>4), col = lane&15 (+N tile)
  const int mbase = row0 + (lane >> 4) * 8;
  const int nloc  = lane & 15;
#pragma unroll
  for (int nt = 0; nt < 2; ++nt) {
    const int col = col0 + nt * 16 + nloc;
    if (col >= nmax) continue;
    const float bb = bias ? bias[col] : 0.0f;
    const v8f acc = nt ? acc1 : acc0;
#pragma unroll
    for (int r = 0; r < 8; ++r) {
      const int row = mbase + r;
      const float v = acc[r] + bb;
      if (Cf) Cf[(size_t)row * crs + (size_t)col * ccs + coff] = v;
      if (Cb) Cb[(size_t)row * cb_ld + col] = f32_to_bf16(v);
    }
  }
}

// ---------------------------------------------------------------------------
// Synchronisation update (both 'action' and 'out' flavors):
//   alpha = r*alpha + act[il]*act[ir];  beta = r*beta + 1;  sy = alpha/sqrt(beta)
// ---------------------------------------------------------------------------
__global__ void __launch_bounds__(256) k_sync(
    const float* __restrict__ act, const int* __restrict__ il,
    const int* __restrict__ ir, const float* __restrict__ decay,
    float* __restrict__ alpha, float* __restrict__ beta,
    float* __restrict__ sy_f, ushort_t* __restrict__ sy_b) {
  const int idx = blockIdx.x * 256 + threadIdx.x;
  if (idx >= B * NSYNC) return;
  const int b = idx / NSYNC;
  const int j = idx - b * NSYNC;
  float dc = decay[j];
  dc = fminf(fmaxf(dc, 0.0f), 15.0f);
  const float r = __expf(-dc);
  const float p = act[(size_t)b * DMODEL + il[j]] * act[(size_t)b * DMODEL + ir[j]];
  const float a  = r * alpha[idx] + p;
  const float be = r * beta[idx] + 1.0f;
  alpha[idx] = a;
  beta[idx]  = be;
  const float sy = a * rsqrtf(be);
  if (sy_f) sy_f[idx] = sy;
  sy_b[idx] = f32_to_bf16(sy);
}

// ---------------------------------------------------------------------------
// Attention for one (b, h): 64 threads. hd = S = 64. Tiny -> VALU + LDS.
// Writes the attention mix as bf16 (input A of the Wo projection).
// ---------------------------------------------------------------------------
__global__ void __launch_bounds__(64) k_attn(
    const float* __restrict__ Qh, const float* __restrict__ Kf,
    const float* __restrict__ Vf, ushort_t* __restrict__ attn_b) {
  const int b = blockIdx.x / HEADS;
  const int h = blockIdx.x % HEADS;
  const int tid = threadIdx.x;
  __shared__ float qs[HD];
  __shared__ float sc[S];
  __shared__ float red[S];

  qs[tid] = Qh[(size_t)b * DATTN + h * HD + tid];
  __syncthreads();

  const float* kr = Kf + ((size_t)(b * S + tid)) * DATTN + h * HD;
  float a = 0.0f;
#pragma unroll 8
  for (int d = 0; d < HD; ++d) a += qs[d] * kr[d];
  a *= 0.125f;  // 1/sqrt(hd)
  sc[tid] = a;
  red[tid] = a;
  __syncthreads();
  for (int off = 32; off > 0; off >>= 1) {
    if (tid < off) red[tid] = fmaxf(red[tid], red[tid + off]);
    __syncthreads();
  }
  const float mx = red[0];
  __syncthreads();
  const float e = __expf(sc[tid] - mx);
  red[tid] = e;
  __syncthreads();
  for (int off = 32; off > 0; off >>= 1) {
    if (tid < off) red[tid] += red[tid + off];
    __syncthreads();
  }
  const float inv = 1.0f / red[0];
  __syncthreads();
  sc[tid] = e * inv;
  __syncthreads();

  float o = 0.0f;
  const float* vb = Vf + (size_t)b * S * DATTN + h * HD + tid;
  for (int s = 0; s < S; ++s) o += sc[s] * vb[(size_t)s * DATTN];
  attn_b[(size_t)b * DATTN + h * HD + tid] = f32_to_bf16(o);
}

// ---------------------------------------------------------------------------
// GLU + LayerNorm + write new state into the trace ring (slot = t % 25).
// One block per batch row; 256 threads x 8 elements.
// ---------------------------------------------------------------------------
__global__ void __launch_bounds__(256) k_glu_ln(
    const float* __restrict__ h, const float* __restrict__ ln_g,
    const float* __restrict__ ln_b, float* __restrict__ trace, int slot) {
  const int b = blockIdx.x;
  const int tid = threadIdx.x;
  __shared__ float red[256];
  float g[8];
  float s = 0.0f;
#pragma unroll
  for (int i = 0; i < 8; ++i) {
    const int d = tid + i * 256;
    const float a1 = h[(size_t)b * H2 + d];
    const float a2 = h[(size_t)b * H2 + DMODEL + d];
    const float v = a1 / (1.0f + __expf(-a2));  // a1 * sigmoid(a2)
    g[i] = v;
    s += v;
  }
  red[tid] = s; __syncthreads();
  for (int off = 128; off > 0; off >>= 1) {
    if (tid < off) red[tid] += red[tid + off];
    __syncthreads();
  }
  const float mu = red[0] * (1.0f / DMODEL);
  __syncthreads();
  float s2 = 0.0f;
#pragma unroll
  for (int i = 0; i < 8; ++i) { const float dd = g[i] - mu; s2 += dd * dd; }
  red[tid] = s2; __syncthreads();
  for (int off = 128; off > 0; off >>= 1) {
    if (tid < off) red[tid] += red[tid + off];
    __syncthreads();
  }
  const float inv = rsqrtf(red[0] * (1.0f / DMODEL) + 1e-5f);
#pragma unroll
  for (int i = 0; i < 8; ++i) {
    const int d = tid + i * 256;
    const float st = (g[i] - mu) * inv * ln_g[d] + ln_b[d];
    trace[((size_t)b * DMODEL + d) * TTR + slot] = st;
  }
}

// ---------------------------------------------------------------------------
// Neuron-level models: act[b,d] = sum_h gelu(trace[b,d,:] . nw1[d,:,h] + nb1) * nw2 + nb2
// Ring-buffer order: logical m -> physical slot (t+1+m) % 25.
// Also refreshes the act half of the 'pre' bf16 buffer.
// ---------------------------------------------------------------------------
__global__ void __launch_bounds__(256) k_nlm(
    const float* __restrict__ trace, const float* __restrict__ nw1,
    const float* __restrict__ nb1, const float* __restrict__ nw2,
    const float* __restrict__ nb2, float* __restrict__ act,
    ushort_t* __restrict__ pre_b, int t) {
  const int idx = blockIdx.x * 256 + threadIdx.x;
  if (idx >= B * DMODEL) return;
  const int b = idx / DMODEL;
  const int d = idx - b * DMODEL;

  float tr[TTR];
  const float* trp = trace + ((size_t)b * DMODEL + d) * TTR;
#pragma unroll
  for (int m = 0; m < TTR; ++m) tr[m] = trp[(t + 1 + m) % TTR];

  const float* w1 = nw1 + (size_t)d * TTR * HNLM;
  const float* b1 = nb1 + (size_t)d * HNLM;
  const float* w2 = nw2 + (size_t)d * HNLM;
  float o = 0.0f;
  for (int hh = 0; hh < HNLM; ++hh) {
    float sacc = b1[hh];
#pragma unroll
    for (int m = 0; m < TTR; ++m) sacc += tr[m] * w1[m * HNLM + hh];
    const float c = 0.7978845608028654f * (sacc + 0.044715f * sacc * sacc * sacc);
    const float gel = 0.5f * sacc * (1.0f + tanhf(c));  // tanh-approx gelu
    o += gel * w2[hh];
  }
  o += nb2[d];
  act[idx] = o;
  pre_b[(size_t)b * PREMW + DATTN + d] = f32_to_bf16(o);
}

// ---------------------------------------------------------------------------
// Entropy / certainty from the raw predictions already stored in d_out.
// ---------------------------------------------------------------------------
__global__ void __launch_bounds__(256) k_cert(
    const float* __restrict__ preds, float* __restrict__ out_cert, int t) {
  const int b = blockIdx.x;
  const int tid = threadIdx.x;
  __shared__ float red[256];
  float x[4];
  int cnt = 0;
  float mx = -3.4e38f;
  for (int o = tid; o < OUT; o += 256) {
    x[cnt] = preds[(size_t)b * OUT * T + (size_t)o * T + t];
    mx = fmaxf(mx, x[cnt]);
    ++cnt;
  }
  red[tid] = mx; __syncthreads();
  for (int off = 128; off > 0; off >>= 1) {
    if (tid < off) red[tid] = fmaxf(red[tid], red[tid + off]);
    __syncthreads();
  }
  mx = red[0]; __syncthreads();
  float se = 0.0f;
  for (int i = 0; i < cnt; ++i) se += __expf(x[i] - mx);
  red[tid] = se; __syncthreads();
  for (int off = 128; off > 0; off >>= 1) {
    if (tid < off) red[tid] += red[tid + off];
    __syncthreads();
  }
  const float lse = mx + __logf(red[0]);
  __syncthreads();
  float ent = 0.0f;
  for (int i = 0; i < cnt; ++i) {
    const float lp = x[i] - lse;
    ent += __expf(lp) * lp;
  }
  red[tid] = ent; __syncthreads();
  for (int off = 128; off > 0; off >>= 1) {
    if (tid < off) red[tid] += red[tid + off];
    __syncthreads();
  }
  if (tid == 0) {
    const float ne = -red[0] * (1.0f / logf((float)OUT));
    out_cert[(size_t)b * 2 * T + t]     = ne;
    out_cert[(size_t)b * 2 * T + T + t] = 1.0f - ne;
  }
}

// ---------------------------------------------------------------------------
// Initialisation: act = start_state (bcast), alphaA/betaA = 0,
// alphaO = pairing(start_state), betaO = 1, trace = start_trace (bcast).
// ---------------------------------------------------------------------------
__global__ void __launch_bounds__(256) k_init_state(
    const float* __restrict__ start_state, const int* __restrict__ ilo,
    const int* __restrict__ iro, float* __restrict__ act,
    ushort_t* __restrict__ pre_b, float* __restrict__ aA, float* __restrict__ bA,
    float* __restrict__ aO, float* __restrict__ bO) {
  const int idx = blockIdx.x * 256 + threadIdx.x;
  if (idx >= B * DMODEL) return;
  const int b = idx / DMODEL;
  const int d = idx - b * DMODEL;
  const float v = start_state[d];
  act[idx] = v;
  pre_b[(size_t)b * PREMW + DATTN + d] = f32_to_bf16(v);
  if (d < NSYNC) {
    const int j = b * NSYNC + d;
    aA[j] = 0.0f;
    bA[j] = 0.0f;
    aO[j] = start_state[ilo[d]] * start_state[iro[d]];
    bO[j] = 1.0f;
  }
}

__global__ void __launch_bounds__(256) k_init_trace(
    const float* __restrict__ start_trace, float* __restrict__ trace) {
  const long idx = (long)blockIdx.x * 256 + threadIdx.x;
  if (idx >= (long)B * DMODEL * TTR) return;
  trace[idx] = start_trace[idx % ((long)DMODEL * TTR)];
}

__global__ void __launch_bounds__(256) k_copyf(const float* __restrict__ src,
                                               float* __restrict__ dst, int n) {
  const int i = blockIdx.x * 256 + threadIdx.x;
  if (i < n) dst[i] = src[i];
}

// ---------------------------------------------------------------------------
// Host orchestration (graph-capture safe; all launches on `stream`).
// ---------------------------------------------------------------------------
extern "C" void kernel_launch(void* const* d_in, const int* in_sizes, int n_in,
                              void* d_out, int out_size, void* d_ws, size_t ws_size,
                              hipStream_t stream) {
  (void)in_sizes; (void)n_in; (void)out_size; (void)ws_size;

  const float* x   = (const float*)d_in[0];
  const float* Wf  = (const float*)d_in[1];
  const float* bf_ = (const float*)d_in[2];
  const float* start_trace = (const float*)d_in[3];
  const float* start_state = (const float*)d_in[4];
  const float* decay_a = (const float*)d_in[5];
  const float* decay_o = (const float*)d_in[6];
  const float* Wqp = (const float*)d_in[7];
  const float* bqp = (const float*)d_in[8];
  const float* Wq  = (const float*)d_in[9];
  const float* bq  = (const float*)d_in[10];
  const float* Wk  = (const float*)d_in[11];
  const float* bk  = (const float*)d_in[12];
  const float* Wv  = (const float*)d_in[13];
  const float* bv  = (const float*)d_in[14];
  const float* Wo  = (const float*)d_in[15];
  const float* bo  = (const float*)d_in[16];
  const float* Ws1 = (const float*)d_in[17];
  const float* bs1 = (const float*)d_in[18];
  const float* ln_g = (const float*)d_in[19];
  const float* ln_b = (const float*)d_in[20];
  const float* nw1 = (const float*)d_in[21];
  const float* nb1 = (const float*)d_in[22];
  const float* nw2 = (const float*)d_in[23];
  const float* nb2 = (const float*)d_in[24];
  const float* Wout = (const float*)d_in[25];
  const float* bout = (const float*)d_in[26];
  const int* idx_la = (const int*)d_in[27];
  const int* idx_ra = (const int*)d_in[28];
  const int* idx_lo = (const int*)d_in[29];
  const int* idx_ro = (const int*)d_in[30];
  float* out = (float*)d_out;

  // Deterministic bump allocator over d_ws (same layout every call).
  size_t off = 0;
  auto alloc = [&](size_t bytes) -> void* {
    void* p = (char*)d_ws + off;
    off += (bytes + 255) & ~(size_t)255;
    return p;
  };
  ushort_t* WfT   = (ushort_t*)alloc((size_t)DATTN * DIN * 2);
  ushort_t* WkT   = (ushort_t*)alloc((size_t)DATTN * DATTN * 2);
  ushort_t* WvT   = (ushort_t*)alloc((size_t)DATTN * DATTN * 2);
  ushort_t* WqpT  = (ushort_t*)alloc((size_t)DATTN * NSYNC * 2);
  ushort_t* WqT   = (ushort_t*)alloc((size_t)DATTN * DATTN * 2);
  ushort_t* WoT   = (ushort_t*)alloc((size_t)DATTN * DATTN * 2);
  ushort_t* Ws1T  = (ushort_t*)alloc((size_t)H2 * PREMW * 2);
  ushort_t* WoutT = (ushort_t*)alloc((size_t)OUTP * NSYNC * 2);
  ushort_t* x_b   = (ushort_t*)alloc((size_t)B * S * DIN * 2);
  ushort_t* kv_b  = (ushort_t*)alloc((size_t)B * S * DATTN * 2);
  float*    Kf    = (float*)alloc((size_t)B * S * DATTN * 4);
  float*    Vf    = (float*)alloc((size_t)B * S * DATTN * 4);
  float*    act   = (float*)alloc((size_t)B * DMODEL * 4);
  float*    trace = (float*)alloc((size_t)B * DMODEL * TTR * 4);
  float*    aA    = (float*)alloc((size_t)B * NSYNC * 4);
  float*    bA    = (float*)alloc((size_t)B * NSYNC * 4);
  float*    aO    = (float*)alloc((size_t)B * NSYNC * 4);
  float*    bO    = (float*)alloc((size_t)B * NSYNC * 4);
  ushort_t* syA_b = (ushort_t*)alloc((size_t)B * NSYNC * 2);
  ushort_t* q_b   = (ushort_t*)alloc((size_t)B * DATTN * 2);
  float*    q2    = (float*)alloc((size_t)B * DATTN * 4);
  ushort_t* attn_b= (ushort_t*)alloc((size_t)B * DATTN * 2);
  ushort_t* pre_b = (ushort_t*)alloc((size_t)B * PREMW * 2);
  float*    hbuf  = (float*)alloc((size_t)B * H2 * 4);
  float*    syO_f = (float*)alloc((size_t)B * NSYNC * 4);
  ushort_t* syO_b = (ushort_t*)alloc((size_t)B * NSYNC * 2);

  auto convt = [&](const float* W, ushort_t* WT, int K, int N, int Npad) {
    const long total = (long)Npad * K;
    k_convT<<<(unsigned)((total + 255) / 256), 256, 0, stream>>>(W, WT, K, N, total);
  };
  auto gemm = [&](const ushort_t* A, const ushort_t* BT, const float* bias,
                  float* Cf, long crs, long ccs, long coff,
                  ushort_t* Cb, int cb_ld, int Mrows, int Ncols, int K, int nmax) {
    dim3 grid((unsigned)(Ncols / 32), (unsigned)(Mrows / 64));
    k_gemm_bf16<<<grid, 128, 0, stream>>>(A, BT, bias, Cf, crs, ccs, coff,
                                          Cb, cb_ld, K, nmax);
  };

  // ---- one-time: weight conversion (f32 -> bf16, N x K layout) ----
  convt(Wf,   WfT,   DIN,   DATTN, DATTN);
  convt(Wk,   WkT,   DATTN, DATTN, DATTN);
  convt(Wv,   WvT,   DATTN, DATTN, DATTN);
  convt(Wqp,  WqpT,  NSYNC, DATTN, DATTN);
  convt(Wq,   WqT,   DATTN, DATTN, DATTN);
  convt(Wo,   WoT,   DATTN, DATTN, DATTN);
  convt(Ws1,  Ws1T,  PREMW, H2,    H2);
  convt(Wout, WoutT, NSYNC, OUT,   OUTP);
  k_conv<<<(unsigned)(((long)B * S * DIN + 255) / 256), 256, 0, stream>>>(
      x, x_b, (long)B * S * DIN);

  // ---- one-time: kv = x@Wf + bf ; K = kv@Wk + bk ; V = kv@Wv + bv ----
  gemm(x_b,  WfT, bf_, nullptr, 0, 0, 0, kv_b, DATTN, B * S, DATTN, DIN,   DATTN);
  gemm(kv_b, WkT, bk,  Kf, DATTN, 1, 0, nullptr, 0,   B * S, DATTN, DATTN, DATTN);
  gemm(kv_b, WvT, bv,  Vf, DATTN, 1, 0, nullptr, 0,   B * S, DATTN, DATTN, DATTN);

  // ---- init recurrent state ----
  k_init_state<<<(B * DMODEL + 255) / 256, 256, 0, stream>>>(
      start_state, idx_lo, idx_ro, act, pre_b, aA, bA, aO, bO);
  k_init_trace<<<(unsigned)(((long)B * DMODEL * TTR + 255) / 256), 256, 0, stream>>>(
      start_trace, trace);

  // ---- T sequential ticks ----
  for (int t = 0; t < T; ++t) {
    // action synchronisation -> syA (bf16)
    k_sync<<<(B * NSYNC + 255) / 256, 256, 0, stream>>>(
        act, idx_la, idx_ra, decay_a, aA, bA, nullptr, syA_b);
    // q = syA@Wqp + bqp (bf16) ; Qh = q@Wq + bq (f32)
    gemm(syA_b, WqpT, bqp, nullptr, 0, 0, 0, q_b, DATTN, B, DATTN, NSYNC, DATTN);
    gemm(q_b,   WqT,  bq,  q2, DATTN, 1, 0, nullptr, 0, B, DATTN, DATTN, DATTN);
    // attention (tiny, VALU) -> attn (bf16)
    k_attn<<<B * HEADS, 64, 0, stream>>>(q2, Kf, Vf, attn_b);
    // ao = attn@Wo + bo  -> cols [0,512) of pre (bf16, ld = 2560)
    gemm(attn_b, WoT, bo, nullptr, 0, 0, 0, pre_b, PREMW, B, DATTN, DATTN, DATTN);
    // h = pre@Ws1 + bs1 (the dominant 64x2560x4096 GEMM, L2-resident bf16)
    gemm(pre_b, Ws1T, bs1, hbuf, H2, 1, 0, nullptr, 0, B, H2, PREMW, H2);
    // GLU + LayerNorm -> new state into trace ring slot t%25
    k_glu_ln<<<B, 256, 0, stream>>>(hbuf, ln_g, ln_b, trace, t % TTR);
    // neuron-level models -> act (also refreshes act half of pre)
    k_nlm<<<(B * DMODEL) / 256, 256, 0, stream>>>(
        trace, nw1, nb1, nw2, nb2, act, pre_b, t);
    // out synchronisation -> syO (f32 + bf16)
    k_sync<<<(B * NSYNC + 255) / 256, 256, 0, stream>>>(
        act, idx_lo, idx_ro, decay_o, aO, bO, syO_f, syO_b);
    // pred = syO@Wout + bout, stored directly into d_out with (B,OUT,T) strides
    gemm(syO_b, WoutT, bout, out, (long)OUT * T, (long)T, (long)t,
         nullptr, 0, B, OUTP, NSYNC, OUT);
    // entropy / certainty from the stored predictions
    k_cert<<<B, 256, 0, stream>>>(out, out + PRED_ELEMS, t);
  }

  // final syO -> tail of d_out
  k_copyf<<<(B * NSYNC + 255) / 256, 256, 0, stream>>>(
      syO_f, out + PRED_ELEMS + CERT_ELEMS, B * NSYNC);
}
